// MLC_quantizer_noun_76553497084148
// MI455X (gfx1250) — matrix operand order repaired and checked by reading, not compiled, not run on hardware
//
#include <hip/hip_runtime.h>
#include <stdint.h>

typedef __attribute__((ext_vector_type(16))) __bf16 v16bf;
typedef __attribute__((ext_vector_type(8)))  __bf16 v8bf;
typedef __attribute__((ext_vector_type(8)))  float  v8f;
typedef __attribute__((ext_vector_type(4)))  unsigned int v4u;
typedef __attribute__((ext_vector_type(8)))  int v8i;
typedef __attribute__((ext_vector_type(4)))  int v4i;

#define N_NODES 8192
#define KDIM    1024
#define EDIM    256
#define NEDGE   131072
#define NF      16384      // rows of zf (64*16*16)
#define NCODE   4096       // codes per codebook half
#define DW      4          // waves per block in distance kernels
#define TILE_B  (16 * EDIM)        // one B tile: 16 codes x 256 bf16 = 8 KB

__device__ __forceinline__ __bf16 f2bf(float x) { return (__bf16)x; }

// A-fragment (16x32 bf16): lane m = lane&15, khalf = lane>>4.
// elems 0..7  = A[m, kb + khalf*8 + i],  elems 8..15 = A[m, kb + 16 + khalf*8 + i]
__device__ __forceinline__ v16bf load_afrag(const __bf16* p, int kh) {
  v8bf lo = *(const v8bf*)(p + kh * 8);
  v8bf hi = *(const v8bf*)(p + 16 + kh * 8);
  v16bf r;
#pragma unroll
  for (int i = 0; i < 8; ++i) { r[i] = lo[i]; r[i + 8] = hi[i]; }
  return r;
}

// B-fragment (32x16 bf16), B stored transposed (N x K row-major):
// lane n = lane&15, khalf = lane>>4; elems 0..7 = Bt[n, kb + khalf*8 + i], elems 8..15 = Bt[n, kb + 16 + khalf*8 + i]
__device__ __forceinline__ v16bf load_bfrag(const __bf16* p, int kh) {
  v8bf lo = *(const v8bf*)(p + kh * 8);
  v8bf hi = *(const v8bf*)(p + 16 + kh * 8);
  v16bf r;
#pragma unroll
  for (int i = 0; i < 8; ++i) { r[i] = lo[i]; r[i + 8] = hi[i]; }
  return r;
}

// ---- Tensor Data Mover: DMA one 16x256-bf16 tile (16 rows x 512 B) into LDS.
// D# group0: count=1 | lds_addr | global_addr(57b) | type=2
// D# group1: data_size=1(2B); tensor_dim0=256, tensor_dim1=16;
//            tile_dim0=256, tile_dim1=16; tensor_dim0_stride=256
__device__ __forceinline__ void tdm_load_tile(const __bf16* gsrc, unsigned int lds_off) {
  uint64_t ga = (uint64_t)(uintptr_t)gsrc;
  v4u g0;
  g0.x = 1u;                                    // count=1, user descriptor
  g0.y = lds_off;                               // lds_addr (bytes)
  g0.z = (unsigned int)ga;                      // global_addr[31:0]
  g0.w = (unsigned int)(ga >> 32) | 0x80000000u; // global_addr[56:32] | type=2
  v8i g1;
  g1[0] = 0x00010000;          // workgroup_mask=0, data_size=1 (2 bytes)
  g1[1] = (int)(256u << 16);   // tensor_dim0 = 256 (bits 63:48)
  g1[2] = (int)(16u << 16);    // tensor_dim0 hi=0 | tensor_dim1 = 16 (bits 95:80)
  g1[3] = (int)(256u << 16);   // tensor_dim1 hi=0 | tile_dim0 = 256 (bits 127:112)
  g1[4] = 16;                  // tile_dim1 = 16, tile_dim2 = 0
  g1[5] = 256;                 // tensor_dim0_stride = 256 (bits 191:160)
  g1[6] = 0;                   // stride0 hi | tensor_dim1_stride lo
  g1[7] = 0;
  v4i gz4 = {0, 0, 0, 0};
  v8i gz8 = {0, 0, 0, 0, 0, 0, 0, 0};
  __builtin_amdgcn_tensor_load_to_lds(g0, g1, gz4, gz4, gz8, 0);
}

// ---------------- Generic NT GEMM: C[M,N] = A[M,K] * Bt[N,K]^T (bf16 in, f32 out)
__global__ void k_wmma_gemm_nt(const __bf16* __restrict__ A, const __bf16* __restrict__ Bt,
                               float* __restrict__ C, int M, int N, int K) {
  const int wave = threadIdx.x >> 5;
  const int lane = threadIdx.x & 31;
  const int nlo  = lane & 15;
  const int kh   = lane >> 4;
  const int tm   = blockIdx.y * (blockDim.x >> 5) + wave;
  const int tn   = blockIdx.x;
  if (tm * 16 >= M) return;
  const __bf16* arow = A  + (size_t)(tm * 16 + nlo) * K;
  const __bf16* brow = Bt + (size_t)(tn * 16 + nlo) * K;
  v8f acc = {};
  for (int kb = 0; kb < K; kb += 32) {
    v16bf af  = load_afrag(arow + kb, kh);
    v16bf bfv = load_bfrag(brow + kb, kh);
    acc = __builtin_amdgcn_wmma_f32_16x16x32_bf16(false, af, false, bfv, (short)0, acc,
                                                  false, false);
  }
  const int m0 = tm * 16 + kh * 8;
  const int cn = tn * 16 + nlo;
#pragma unroll
  for (int r = 0; r < 8; ++r) C[(size_t)(m0 + r) * N + cn] = acc[r];
}

// ---------------- Fused distance + top-4 (adjective codebook), TDM-staged B
__global__ void k_dist_top4(const __bf16* __restrict__ zbf, const float* __restrict__ znorm,
                            const __bf16* __restrict__ ebf, const float* __restrict__ enorm,
                            int* __restrict__ idx_ws, float* __restrict__ out_idx) {
  __shared__ __bf16 bstage[2 * TILE_B];          // double-buffered B tile (2 x 8 KB)
  __shared__ float sd[DW][16][16][4];
  __shared__ int   si[DW][16][16][4];
  const int wave  = threadIdx.x >> 5;
  const int lane  = threadIdx.x & 31;
  const int nlo   = lane & 15;
  const int kh    = lane >> 4;
  const int fbase = (blockIdx.x * DW + wave) * 16;
  const unsigned int lds_base = (unsigned int)(uintptr_t)&bstage[0];

  v16bf af[8];
  const __bf16* arow = zbf + (size_t)(fbase + nlo) * EDIM;
#pragma unroll
  for (int kt = 0; kt < 8; ++kt) af[kt] = load_afrag(arow + kt * 32, kh);

  float zn[8];
#pragma unroll
  for (int r = 0; r < 8; ++r) zn[r] = znorm[fbase + kh * 8 + r];

  float d0[8], d1[8], d2[8], d3[8];
  int   i0[8], i1[8], i2[8], i3[8];
#pragma unroll
  for (int r = 0; r < 8; ++r) {
    d0[r] = d1[r] = d2[r] = d3[r] = 3.0e38f;
    i0[r] = i1[r] = i2[r] = i3[r] = 0;
  }

  if (wave == 0) tdm_load_tile(ebf, lds_base);   // prefetch tile 0

  for (int nt = 0; nt < NCODE / 16; ++nt) {
    if (wave == 0) __builtin_amdgcn_s_wait_tensorcnt(0);   // tile nt resident
    __syncthreads();                                       // all waves: tile ready,
                                                           // prev tile fully consumed
    if (wave == 0 && nt + 1 < NCODE / 16)
      tdm_load_tile(ebf + (size_t)(nt + 1) * TILE_B,
                    lds_base + (unsigned int)(((nt + 1) & 1) * TILE_B * 2));

    const __bf16* btile = bstage + (nt & 1) * TILE_B + (size_t)nlo * EDIM;
    const int n = nt * 16 + nlo;
    const float en = enorm[n];
    v8f acc = {};
#pragma unroll
    for (int kt = 0; kt < 8; ++kt) {
      v16bf bfv = load_bfrag(btile + kt * 32, kh);         // ds_load_b128 x2
      acc = __builtin_amdgcn_wmma_f32_16x16x32_bf16(false, af[kt], false, bfv, (short)0, acc,
                                                    false, false);
    }
#pragma unroll
    for (int r = 0; r < 8; ++r) {
      float d = zn[r] + en - 2.0f * acc[r];
      if (d < d3[r]) {
        if (d < d2[r]) {
          d3[r] = d2[r]; i3[r] = i2[r];
          if (d < d1[r]) {
            d2[r] = d1[r]; i2[r] = i1[r];
            if (d < d0[r]) { d1[r] = d0[r]; i1[r] = i0[r]; d0[r] = d; i0[r] = n; }
            else           { d1[r] = d; i1[r] = n; }
          } else { d2[r] = d; i2[r] = n; }
        } else { d3[r] = d; i3[r] = n; }
      }
    }
  }

#pragma unroll
  for (int r = 0; r < 8; ++r) {
    const int row = kh * 8 + r;
    sd[wave][row][nlo][0] = d0[r]; sd[wave][row][nlo][1] = d1[r];
    sd[wave][row][nlo][2] = d2[r]; sd[wave][row][nlo][3] = d3[r];
    si[wave][row][nlo][0] = i0[r]; si[wave][row][nlo][1] = i1[r];
    si[wave][row][nlo][2] = i2[r]; si[wave][row][nlo][3] = i3[r];
  }
  __syncthreads();

  if (threadIdx.x < DW * 16) {
    const int w   = threadIdx.x >> 4;
    const int row = threadIdx.x & 15;
    float cd0 = 3.0e38f, cd1 = 3.0e38f, cd2 = 3.0e38f, cd3 = 3.0e38f;
    int   ci0 = 0x7fffffff, ci1 = 0x7fffffff, ci2 = 0x7fffffff, ci3 = 0x7fffffff;
    for (int sl = 0; sl < 16; ++sl) {
#pragma unroll
      for (int j = 0; j < 4; ++j) {
        float d  = sd[w][row][sl][j];
        int   ii = si[w][row][sl][j];
        if ((d < cd3) || (d == cd3 && ii < ci3)) {
          if ((d < cd2) || (d == cd2 && ii < ci2)) {
            cd3 = cd2; ci3 = ci2;
            if ((d < cd1) || (d == cd1 && ii < ci1)) {
              cd2 = cd1; ci2 = ci1;
              if ((d < cd0) || (d == cd0 && ii < ci0)) {
                cd1 = cd0; ci1 = ci0; cd0 = d; ci0 = ii;
              } else { cd1 = d; ci1 = ii; }
            } else { cd2 = d; ci2 = ii; }
          } else { cd3 = d; ci3 = ii; }
        }
      }
    }
    const int f  = blockIdx.x * DW * 16 + w * 16 + row;
    const int b  = f >> 8;
    const int hw = f & 255;
    idx_ws[f * 4 + 0] = ci0; idx_ws[f * 4 + 1] = ci1;
    idx_ws[f * 4 + 2] = ci2; idx_ws[f * 4 + 3] = ci3;
    float* op = out_idx + (size_t)b * 1280 + hw * 4;
    op[0] = (float)ci0; op[1] = (float)ci1; op[2] = (float)ci2; op[3] = (float)ci3;
  }
}

// ---------------- Fused distance + argmin (noun codebook), TDM-staged B
__global__ void k_dist_argmin(const __bf16* __restrict__ zbf, const float* __restrict__ znorm,
                              const __bf16* __restrict__ ebf, const float* __restrict__ enorm,
                              int* __restrict__ idx_ws, float* __restrict__ out_idx) {
  __shared__ __bf16 bstage[2 * TILE_B];
  __shared__ float sd[DW][16][16];
  __shared__ int   si[DW][16][16];
  const int wave  = threadIdx.x >> 5;
  const int lane  = threadIdx.x & 31;
  const int nlo   = lane & 15;
  const int kh    = lane >> 4;
  const int fbase = (blockIdx.x * DW + wave) * 16;
  const unsigned int lds_base = (unsigned int)(uintptr_t)&bstage[0];

  v16bf af[8];
  const __bf16* arow = zbf + (size_t)(fbase + nlo) * EDIM;
#pragma unroll
  for (int kt = 0; kt < 8; ++kt) af[kt] = load_afrag(arow + kt * 32, kh);

  float zn[8];
#pragma unroll
  for (int r = 0; r < 8; ++r) zn[r] = znorm[fbase + kh * 8 + r];

  float bd[8]; int bi[8];
#pragma unroll
  for (int r = 0; r < 8; ++r) { bd[r] = 3.0e38f; bi[r] = 0; }

  if (wave == 0) tdm_load_tile(ebf, lds_base);

  for (int nt = 0; nt < NCODE / 16; ++nt) {
    if (wave == 0) __builtin_amdgcn_s_wait_tensorcnt(0);
    __syncthreads();
    if (wave == 0 && nt + 1 < NCODE / 16)
      tdm_load_tile(ebf + (size_t)(nt + 1) * TILE_B,
                    lds_base + (unsigned int)(((nt + 1) & 1) * TILE_B * 2));

    const __bf16* btile = bstage + (nt & 1) * TILE_B + (size_t)nlo * EDIM;
    const int n = nt * 16 + nlo;
    const float en = enorm[n];
    v8f acc = {};
#pragma unroll
    for (int kt = 0; kt < 8; ++kt) {
      v16bf bfv = load_bfrag(btile + kt * 32, kh);
      acc = __builtin_amdgcn_wmma_f32_16x16x32_bf16(false, af[kt], false, bfv, (short)0, acc,
                                                    false, false);
    }
#pragma unroll
    for (int r = 0; r < 8; ++r) {
      float d = zn[r] + en - 2.0f * acc[r];
      if (d < bd[r]) { bd[r] = d; bi[r] = n; }
    }
  }

#pragma unroll
  for (int r = 0; r < 8; ++r) {
    const int row = kh * 8 + r;
    sd[wave][row][nlo] = bd[r];
    si[wave][row][nlo] = bi[r];
  }
  __syncthreads();

  if (threadIdx.x < DW * 16) {
    const int w   = threadIdx.x >> 4;
    const int row = threadIdx.x & 15;
    float cd = 3.0e38f; int ci = 0x7fffffff;
    for (int sl = 0; sl < 16; ++sl) {
      float d  = sd[w][row][sl];
      int   ii = si[w][row][sl];
      if ((d < cd) || (d == cd && ii < ci)) { cd = d; ci = ii; }
    }
    const int f  = blockIdx.x * DW * 16 + w * 16 + row;
    const int b  = f >> 8;
    const int hw = f & 255;
    idx_ws[f] = ci;
    out_idx[(size_t)b * 1280 + 1024 + hw] = (float)ci;
  }
}

// ---------------- Elementwise / GCN helpers
__global__ void k_cvt_bf16(const float* __restrict__ s, __bf16* __restrict__ d, int n) {
  int i = blockIdx.x * blockDim.x + threadIdx.x;
  if (i < n) d[i] = f2bf(s[i]);
}

__global__ void k_transpose_w(const float* __restrict__ W, __bf16* __restrict__ Wt, int K, int N) {
  int t = blockIdx.x * blockDim.x + threadIdx.x;
  if (t < K * N) {
    int n = t / K, k = t % K;
    Wt[(size_t)n * K + k] = f2bf(W[(size_t)k * N + n]);
  }
}

__global__ void k_deg_init(float* deg) {
  int i = blockIdx.x * blockDim.x + threadIdx.x;
  if (i < N_NODES) deg[i] = 1.0f;          // self loop
}

__global__ void k_deg_count(const int* __restrict__ dst, float* deg) {
  int e = blockIdx.x * blockDim.x + threadIdx.x;
  if (e < NEDGE) atomicAdd(&deg[dst[e]], 1.0f);
}

__global__ void k_dinv(const float* __restrict__ deg, float* __restrict__ dinv) {
  int i = blockIdx.x * blockDim.x + threadIdx.x;
  if (i < N_NODES) dinv[i] = rsqrtf(fmaxf(deg[i], 1e-12f));
}

__global__ void k_agg_init(const float* __restrict__ XW, const float* __restrict__ dinv,
                           const float* __restrict__ bias, float* __restrict__ outb) {
  int i = blockIdx.x * blockDim.x + threadIdx.x;   // N_NODES*EDIM
  int n = i >> 8, c = i & 255;
  float di = dinv[n];
  outb[i] = bias[c] + di * di * XW[i];             // self-loop + bias
}

__global__ void k_agg_edges(const int* __restrict__ src, const int* __restrict__ dst,
                            const float* __restrict__ dinv, const float* __restrict__ XW,
                            float* __restrict__ outb) {
  int e = blockIdx.x;
  int c = threadIdx.x;
  int s = src[e], d = dst[e];
  float nrm = dinv[s] * dinv[d];
  atomicAdd(&outb[(size_t)d * EDIM + c], nrm * XW[(size_t)s * EDIM + c]);
}

__global__ void k_relu_cvt(const float* __restrict__ s, __bf16* __restrict__ d, int n) {
  int i = blockIdx.x * blockDim.x + threadIdx.x;
  if (i < n) d[i] = f2bf(fmaxf(s[i], 0.0f));
}

__global__ void k_row_norm_cvt(const float* __restrict__ s, __bf16* __restrict__ d,
                               float* __restrict__ norms) {
  __shared__ float red[256];
  int row = blockIdx.x, c = threadIdx.x;
  float v = s[(size_t)row * EDIM + c];
  d[(size_t)row * EDIM + c] = f2bf(v);
  red[c] = v * v; __syncthreads();
  for (int st = 128; st > 0; st >>= 1) { if (c < st) red[c] += red[c + st]; __syncthreads(); }
  if (c == 0) norms[row] = red[0];
}

// zf[f,c] = z[b, coff+c, hw] with f = b*256 + hw
__global__ void k_z_prep(const float* __restrict__ z, int coff, __bf16* __restrict__ zbf,
                         float* __restrict__ znorm) {
  __shared__ float red[256];
  int f = blockIdx.x, c = threadIdx.x;
  int b = f >> 8, hw = f & 255;
  float v = z[((size_t)b * 512 + coff + c) * 256 + hw];
  zbf[(size_t)f * EDIM + c] = f2bf(v);
  red[c] = v * v; __syncthreads();
  for (int st = 128; st > 0; st >>= 1) { if (c < st) red[c] += red[c + st]; __syncthreads(); }
  if (c == 0) znorm[f] = red[0];
}

__global__ void k_zero1(float* p) { p[0] = 0.0f; }

__global__ void k_quant_adj(const float* __restrict__ tew, const int* __restrict__ idx1,
                            const float* __restrict__ z, float* __restrict__ out,
                            float* __restrict__ mse) {
  __shared__ float red[256];
  int f = blockIdx.x, c = threadIdx.x;
  int b = f >> 8, hw = f & 255;
  const int* ip = idx1 + f * 4;
  float q = 0.25f * (tew[(size_t)ip[0] * EDIM + c] + tew[(size_t)ip[1] * EDIM + c] +
                     tew[(size_t)ip[2] * EDIM + c] + tew[(size_t)ip[3] * EDIM + c]);
  size_t zo = ((size_t)b * 512 + c) * 256 + hw;
  float zv = z[zo];
  out[zo] = q;                     // straight-through forward value
  float df = q - zv;
  red[c] = df * df; __syncthreads();
  for (int st = 128; st > 0; st >>= 1) { if (c < st) red[c] += red[c + st]; __syncthreads(); }
  if (c == 0) atomicAdd(mse, red[0]);
}

__global__ void k_quant_noun(const float* __restrict__ tew, const int* __restrict__ idx2,
                             const float* __restrict__ z, float* __restrict__ out,
                             float* __restrict__ mse) {
  __shared__ float red[256];
  int f = blockIdx.x, c = threadIdx.x;
  int b = f >> 8, hw = f & 255;
  int ii = idx2[f];
  float q = tew[(size_t)(NCODE + ii) * EDIM + c];
  size_t zo = ((size_t)b * 512 + 256 + c) * 256 + hw;
  float zv = z[zo];
  out[zo] = q;
  float df = q - zv;
  red[c] = df * df; __syncthreads();
  for (int st = 128; st > 0; st >>= 1) { if (c < st) red[c] += red[c + st]; __syncthreads(); }
  if (c == 0) atomicAdd(mse, red[0]);
}

__global__ void k_finalize(const float* __restrict__ mse, float* __restrict__ out) {
  float quan   = mse[0] * (1.0f / 4194304.0f);   // (sumA+sumN) / (64*256*16*16)
  float commit = 0.25f * quan;
  out[8388608] = quan + commit;   // loss
  out[8388609] = commit;          // commit_loss
  out[8388610] = quan;            // quan_loss
}

extern "C" void kernel_launch(void* const* d_in, const int* in_sizes, int n_in,
                              void* d_out, int out_size, void* d_ws, size_t ws_size,
                              hipStream_t stream) {
  (void)in_sizes; (void)n_in; (void)out_size; (void)ws_size;
  const float* zin    = (const float*)d_in[0];
  const float* node_x = (const float*)d_in[1];
  const int*   eidx   = (const int*)d_in[2];
  const float* W1     = (const float*)d_in[3];
  const float* b1     = (const float*)d_in[4];
  const float* W2     = (const float*)d_in[5];
  const float* b2     = (const float*)d_in[6];
  float* out = (float*)d_out;
  char*  ws  = (char*)d_ws;

  // workspace layout (with reuse; ~51 MB total)
  const size_t OFF_ABF   = 0;                       // 16 MB node_x bf16 (reused below)
  const size_t OFF_ZBFA  = 0;                       // 8 MB  (after gemm1)
  const size_t OFF_ZBFN  = 8388608;                 // 8 MB
  const size_t OFF_XW    = 16777216;                // 8 MB  (reused after agg1)
  const size_t OFF_ZNA   = 16777216;                // 64 KB
  const size_t OFF_ZNN   = 16777216 + 65536;        // 64 KB
  const size_t OFF_IDX1  = 16777216 + 131072;       // 256 KB
  const size_t OFF_IDX2  = 16777216 + 393216;       // 64 KB
  const size_t OFF_MSE   = 16777216 + 458752;       // 256 B
  const size_t OFF_OUT1  = 25165824;                // 8 MB  (conv1 out, then HW)
  const size_t OFF_HBF   = 33554432;                // 4 MB
  const size_t OFF_TEW   = 37748736;                // 8 MB
  const size_t OFF_TEWBF = 46137344;                // 4 MB
  const size_t OFF_W1T   = 50331648;                // 512 KB
  const size_t OFF_W2T   = 50855936;                // 128 KB
  const size_t OFF_DEG   = 50987008;                // 32 KB
  const size_t OFF_DINV  = 51019776;                // 32 KB
  const size_t OFF_ENORM = 51052544;                // 32 KB

  __bf16* Abf   = (__bf16*)(ws + OFF_ABF);
  __bf16* zbfa  = (__bf16*)(ws + OFF_ZBFA);
  __bf16* zbfn  = (__bf16*)(ws + OFF_ZBFN);
  float*  XW    = (float*)(ws + OFF_XW);
  float*  zna   = (float*)(ws + OFF_ZNA);
  float*  znn   = (float*)(ws + OFF_ZNN);
  int*    idx1  = (int*)(ws + OFF_IDX1);
  int*    idx2  = (int*)(ws + OFF_IDX2);
  float*  mse   = (float*)(ws + OFF_MSE);
  float*  out1  = (float*)(ws + OFF_OUT1);
  float*  HW    = (float*)(ws + OFF_OUT1);
  __bf16* hbf   = (__bf16*)(ws + OFF_HBF);
  float*  tew   = (float*)(ws + OFF_TEW);
  __bf16* tewbf = (__bf16*)(ws + OFF_TEWBF);
  __bf16* W1t   = (__bf16*)(ws + OFF_W1T);
  __bf16* W2t   = (__bf16*)(ws + OFF_W2T);
  float*  deg   = (float*)(ws + OFF_DEG);
  float*  dinv  = (float*)(ws + OFF_DINV);
  float*  enorm = (float*)(ws + OFF_ENORM);

  const int* esrc = eidx;
  const int* edst = eidx + NEDGE;

  // ---- GCN conv 1
  k_cvt_bf16<<<(N_NODES * KDIM) / 256, 256, 0, stream>>>(node_x, Abf, N_NODES * KDIM);
  k_transpose_w<<<(KDIM * EDIM) / 256, 256, 0, stream>>>(W1, W1t, KDIM, EDIM);
  k_transpose_w<<<(EDIM * EDIM) / 256, 256, 0, stream>>>(W2, W2t, EDIM, EDIM);
  k_deg_init<<<N_NODES / 256, 256, 0, stream>>>(deg);
  k_deg_count<<<NEDGE / 256, 256, 0, stream>>>(edst, deg);
  k_dinv<<<N_NODES / 256, 256, 0, stream>>>(deg, dinv);

  dim3 gg(EDIM / 16, N_NODES / (16 * 8));
  k_wmma_gemm_nt<<<gg, 256, 0, stream>>>(Abf, W1t, XW, N_NODES, EDIM, KDIM);
  k_agg_init<<<(N_NODES * EDIM) / 256, 256, 0, stream>>>(XW, dinv, b1, out1);
  k_agg_edges<<<NEDGE, 256, 0, stream>>>(esrc, edst, dinv, XW, out1);
  k_relu_cvt<<<(N_NODES * EDIM) / 256, 256, 0, stream>>>(out1, hbf, N_NODES * EDIM);

  // ---- GCN conv 2 (HW overwrites out1 region; out1 fully consumed into hbf)
  k_wmma_gemm_nt<<<gg, 256, 0, stream>>>(hbf, W2t, HW, N_NODES, EDIM, EDIM);
  k_agg_init<<<(N_NODES * EDIM) / 256, 256, 0, stream>>>(HW, dinv, b2, tew);
  k_agg_edges<<<NEDGE, 256, 0, stream>>>(esrc, edst, dinv, HW, tew);

  // ---- VQ prep: bf16 copies + squared row norms
  k_row_norm_cvt<<<N_NODES, 256, 0, stream>>>(tew, tewbf, enorm);
  k_z_prep<<<NF, 256, 0, stream>>>(zin, 0, zbfa, zna);
  k_z_prep<<<NF, 256, 0, stream>>>(zin, 256, zbfn, znn);
  k_zero1<<<1, 1, 0, stream>>>(mse);

  // ---- fused distance GEMM + selection (TDM-staged codebook tiles)
  float* out_idx = out + 8388611;
  k_dist_top4<<<NF / (DW * 16), DW * 32, 0, stream>>>(zbfa, zna, tewbf, enorm, idx1, out_idx);
  k_dist_argmin<<<NF / (DW * 16), DW * 32, 0, stream>>>(zbfn, znn,
                                                        tewbf + (size_t)NCODE * EDIM,
                                                        enorm + NCODE, idx2, out_idx);

  // ---- gather quantized vectors, write z_q, accumulate MSE
  k_quant_adj<<<NF, 256, 0, stream>>>(tew, idx1, zin, out, mse);
  k_quant_noun<<<NF, 256, 0, stream>>>(tew, idx2, zin, out, mse);
  k_finalize<<<1, 1, 0, stream>>>(mse, out);
}
